// GroupedQueryAttention_61031485276747
// MI455X (gfx1250) — matrix operand order repaired and checked by reading, not compiled, not run on hardware
//
#include <hip/hip_runtime.h>

typedef unsigned short bfr;   // raw bf16 bits
typedef unsigned int   u32;

typedef __attribute__((ext_vector_type(16))) __bf16 v16bf;
typedef __attribute__((ext_vector_type(8)))  float  v8f;

union Frag { v16bf v; u32 u[8]; uint4 qq[2]; };

#define HID   2048
#define SEQ   2048
#define NB    2
#define NQH   32
#define NKVH  8
#define HD    64
#define GRP   4
#define KVD   (NKVH * HD)     // 512

__device__ __forceinline__ bfr f2bf(float f) {
  u32 u = __float_as_uint(f);
  u32 r = u + 0x7FFFu + ((u >> 16) & 1u);
  return (bfr)(r >> 16);
}

// ---------------------------------------------------------------- convert
__global__ void cvt_f32_bf16(const float* __restrict__ src, bfr* __restrict__ dst, int n) {
  int i = blockIdx.x * blockDim.x + threadIdx.x;
  if (i < n) dst[i] = f2bf(src[i]);
}

// ---------------------------------------------------------------- GEMM
// C[M,N] = A[M,K] (bf16 rm) x W[N,K]^T (bf16 rm) + bias
// mode 0: bf16 row-major   mode 1: f32 row-major
// mode 2: bf16 transposed per batch: out[(b*N + col)*Lb + l], row = b*Lb + l
#define BM 128
#define BN 128
#define KT 32

__global__ __launch_bounds__(256) void gemm_bf16(const bfr* __restrict__ A,
                                                 const bfr* __restrict__ W,
                                                 const float* __restrict__ bias,
                                                 void* __restrict__ outp,
                                                 int M, int N, int K, int mode, int Lb) {
  __shared__ bfr sA[BM][KT];   // 8 KB
  __shared__ bfr sB[BN][KT];   // 8 KB
  const int m0   = blockIdx.x * BM;
  const int n0   = blockIdx.y * BN;
  const int tid  = threadIdx.x;
  const int lane = tid & 31;
  const int wave = tid >> 5;        // 8 waves
  const int wm   = wave & 3;        // 32-row strip
  const int wn   = wave >> 2;       // 64-col strip
  const int hl   = lane >> 4;
  const int ln   = lane & 15;

  // staging: thread covers rows (srow, srow+64), one 8-elem quad each
  const int srow  = tid >> 2;
  const int soff  = (tid & 3) * 8;
  const bfr* gA = A + (size_t)(m0 + srow) * K + soff;
  const bfr* gB = W + (size_t)(n0 + srow) * K + soff;
  const size_t rstep = (size_t)64 * K;

  v8f acc[2][4];
  for (int s = 0; s < 2; ++s)
    for (int c = 0; c < 4; ++c)
      for (int r = 0; r < 8; ++r) acc[s][c][r] = 0.0f;

  for (int k0 = 0; k0 < K; k0 += KT) {
    // issue all 4 global loads first, then LDS stores (keeps loads in flight)
    uint4 a0 = *(const uint4*)(gA + k0);
    uint4 a1 = *(const uint4*)(gA + rstep + k0);
    uint4 b0 = *(const uint4*)(gB + k0);
    uint4 b1 = *(const uint4*)(gB + rstep + k0);
    *(uint4*)&sA[srow     ][soff] = a0;
    *(uint4*)&sA[srow + 64][soff] = a1;
    *(uint4*)&sB[srow     ][soff] = b0;
    *(uint4*)&sB[srow + 64][soff] = b1;
    __syncthreads();

    // A fragments (16x32): elem 0..7 -> K=hl*8+i ; elem 8..15 -> K=16+hl*8+(i-8)
    Frag a[2];
    #pragma unroll
    for (int s = 0; s < 2; ++s) {
      const bfr* arow = &sA[wm * 32 + s * 16 + ln][0];
      a[s].qq[0] = *(const uint4*)(arow + hl * 8);
      a[s].qq[1] = *(const uint4*)(arow + 16 + hl * 8);
    }
    #pragma unroll
    for (int c = 0; c < 4; ++c) {
      // B fragment (32x16): elem 0..7 -> K=hl*8+i ; elem 8..15 -> K=16+hl*8+(i-8)
      Frag bfrag;
      const bfr* brow = &sB[wn * 64 + c * 16 + ln][0];
      bfrag.qq[0] = *(const uint4*)(brow + hl * 8);
      bfrag.qq[1] = *(const uint4*)(brow + 16 + hl * 8);
      #pragma unroll
      for (int s = 0; s < 2; ++s)
        acc[s][c] = __builtin_amdgcn_wmma_f32_16x16x32_bf16(
            false, a[s].v, false, bfrag.v, (short)0, acc[s][c], false, false);
    }
    __syncthreads();
  }

  // C layout: row = r + 8*hl, col = ln
  for (int s = 0; s < 2; ++s) {
    for (int c = 0; c < 4; ++c) {
      int col = n0 + wn * 64 + c * 16 + ln;
      float bv = bias[col];
      for (int r = 0; r < 8; ++r) {
        int row = m0 + wm * 32 + s * 16 + r + 8 * hl;
        float val = acc[s][c][r] + bv;
        if (mode == 1) {
          ((float*)outp)[(size_t)row * N + col] = val;
        } else if (mode == 0) {
          ((bfr*)outp)[(size_t)row * N + col] = f2bf(val);
        } else {
          int b = row / Lb, l = row - b * Lb;
          ((bfr*)outp)[((size_t)b * N + col) * Lb + l] = f2bf(val);
        }
      }
    }
  }
}

// ---------------------------------------------------------------- attention
// Block = 4 waves = the 4 grouped q-heads of one KV head, same 16 query rows.
// K/V tiles staged once in LDS and shared by all 4 waves (4x cut in L2 traffic).
__global__ __launch_bounds__(128) void attn_kernel(const bfr* __restrict__ qb,
                                                   const bfr* __restrict__ kb,
                                                   const bfr* __restrict__ vT,
                                                   const int* __restrict__ amask,
                                                   bfr* __restrict__ ab) {
  __shared__ bfr sK[32][64];          // keys j0..j0+31, dims 0..63   (4 KB)
  __shared__ bfr sV[64][32];          // dims 0..63, keys j0..j0+31   (4 KB)
  __shared__ bfr sP[4][16][32];       // per-wave P bounce             (4 KB)
  const int q0   = blockIdx.x * 16;
  const int bh   = blockIdx.y;        // 0..NB*NKVH-1
  const int b    = bh >> 3;
  const int hkv  = bh & 7;
  const int tid  = threadIdx.x;
  const int wave = tid >> 5;
  const int lane = tid & 31;
  const int hl   = lane >> 4;
  const int ln   = lane & 15;
  const int hq   = hkv * GRP + wave;  // this wave's q-head

  // staging bases: thread covers K rows (skey, skey+16) and V rows (sfeat, sfeat+32)
  const int skey  = tid >> 3;
  const int koff  = (tid & 7) * 8;
  const int sfeat = tid >> 2;
  const int voff  = (tid & 3) * 8;
  const bfr* gK = kb + ((size_t)(b * SEQ + skey)) * KVD + hkv * HD + koff;
  const bfr* gV = vT + ((size_t)(b * KVD) + hkv * HD + sfeat) * SEQ + voff;
  const size_t kstep = (size_t)16 * KVD;   // +16 keys
  const size_t vstep = (size_t)32 * SEQ;   // +32 feats

  // Q fragments: A-layout, two 32-dim k-steps
  Frag qf[2];
  {
    const bfr* qrow = qb + ((size_t)(b * SEQ + q0 + ln)) * HID + hq * HD;
    #pragma unroll
    for (int s = 0; s < 2; ++s) {
      qf[s].qq[0] = *(const uint4*)(qrow + s * 32 + hl * 8);
      qf[s].qq[1] = *(const uint4*)(qrow + s * 32 + 16 + hl * 8);
    }
  }

  float mrow[8], lrow[8];
  v8f oacc[4];
  for (int r = 0; r < 8; ++r) { mrow[r] = -1e30f; lrow[r] = 0.0f; }
  for (int c = 0; c < 4; ++c)
    for (int r = 0; r < 8; ++r) oacc[c][r] = 0.0f;

  const int jend = q0 + 16;                      // causal: keys < jend
  for (int j0 = 0; j0 < jend; j0 += 32) {
    // ---- stage K tile [32 keys][64 dims] and V tile [64 dims][32 keys]
    uint4 k0v = *(const uint4*)(gK + (size_t)j0 * KVD);
    uint4 k1v = *(const uint4*)(gK + (size_t)j0 * KVD + kstep);
    uint4 v0v = *(const uint4*)(gV + j0);
    uint4 v1v = *(const uint4*)(gV + vstep + j0);
    *(uint4*)&sK[skey      ][koff] = k0v;
    *(uint4*)&sK[skey + 16 ][koff] = k1v;
    *(uint4*)&sV[sfeat     ][voff] = v0v;
    *(uint4*)&sV[sfeat + 32][voff] = v1v;
    __syncthreads();

    // ---- scores: Q[16,64] x K^T -> two 16x16 tiles
    v8f sc[2];
    for (int t = 0; t < 2; ++t)
      for (int r = 0; r < 8; ++r) sc[t][r] = 0.0f;
    #pragma unroll
    for (int t = 0; t < 2; ++t) {
      #pragma unroll
      for (int s = 0; s < 2; ++s) {
        Frag kf;   // B-layout: elem 0..7 -> dim = s*32 + hl*8 + i ; elem 8..15 -> dim = s*32 + 16 + hl*8 + (i-8)
        const bfr* krow = &sK[t * 16 + ln][s * 32];
        kf.qq[0] = *(const uint4*)(krow + hl * 8);
        kf.qq[1] = *(const uint4*)(krow + 16 + hl * 8);
        sc[t] = __builtin_amdgcn_wmma_f32_16x16x32_bf16(
            false, qf[s].v, false, kf.v, (short)0, sc[t], false, false);
      }
    }

    // ---- scale + causal + padding mask (C-layout: row=r+8*hl, col=ln)
    for (int t = 0; t < 2; ++t) {
      int key = j0 + t * 16 + ln;
      int pad = amask[b * SEQ + key];
      for (int r = 0; r < 8; ++r) {
        int qr = q0 + r + 8 * hl;
        float v = sc[t][r] * 0.125f;               // 1/sqrt(64)
        if (key > qr || pad == 0) v = -1e30f;
        sc[t][r] = v;
      }
    }

    // ---- online softmax (row reductions across 16-lane halves)
    float alpha[8];
    for (int r = 0; r < 8; ++r) {
      float tm = fmaxf(sc[0][r], sc[1][r]);
      for (int off = 1; off < 16; off <<= 1)
        tm = fmaxf(tm, __shfl_xor(tm, off, 32));
      float mnew = fmaxf(mrow[r], tm);
      alpha[r] = __expf(mrow[r] - mnew);
      mrow[r]  = mnew;
    }
    for (int t = 0; t < 2; ++t)
      for (int r = 0; r < 8; ++r)
        sc[t][r] = __expf(sc[t][r] - mrow[r]);
    for (int r = 0; r < 8; ++r) {
      float rs = sc[0][r] + sc[1][r];
      for (int off = 1; off < 16; off <<= 1)
        rs += __shfl_xor(rs, off, 32);
      lrow[r] = lrow[r] * alpha[r] + rs;
    }
    for (int c = 0; c < 4; ++c)
      for (int r = 0; r < 8; ++r) oacc[c][r] *= alpha[r];

    // ---- P (C-layout) -> per-wave LDS -> A-layout fragment
    for (int t = 0; t < 2; ++t)
      for (int r = 0; r < 8; ++r)
        sP[wave][r + 8 * hl][t * 16 + ln] = f2bf(sc[t][r]);
    // per-wave region + in-order DS within a wave: no block barrier needed here
    Frag pf;
    const bfr* prow = &sP[wave][ln][0];
    pf.qq[0] = *(const uint4*)(prow + hl * 8);
    pf.qq[1] = *(const uint4*)(prow + 16 + hl * 8);

    // ---- O += P[16,32] x V[32,64]
    #pragma unroll
    for (int c = 0; c < 4; ++c) {
      Frag vf;   // B-layout: elem 0..7 -> key = hl*8 + i ; elem 8..15 -> key = 16 + hl*8 + (i-8)
      const bfr* vrow = &sV[c * 16 + ln][0];
      vf.qq[0] = *(const uint4*)(vrow + hl * 8);
      vf.qq[1] = *(const uint4*)(vrow + 16 + hl * 8);
      oacc[c] = __builtin_amdgcn_wmma_f32_16x16x32_bf16(
          false, pf.v, false, vf.v, (short)0, oacc[c], false, false);
    }
    __syncthreads();   // protect sK/sV before next iteration's staging
  }

  // ---- normalize + store bf16 attention output [B*L][HID]
  for (int c = 0; c < 4; ++c) {
    int d = c * 16 + ln;
    for (int r = 0; r < 8; ++r) {
      int m = q0 + r + 8 * hl;
      float val = oacc[c][r] / lrow[r];
      ab[((size_t)(b * SEQ + m)) * HID + hq * HD + d] = f2bf(val);
    }
  }
}

// ---------------------------------------------------------------- launch
extern "C" void kernel_launch(void* const* d_in, const int* in_sizes, int n_in,
                              void* d_out, int out_size, void* d_ws, size_t ws_size,
                              hipStream_t stream) {
  const float* x    = (const float*)d_in[0];
  const int*   am   = (const int*)d_in[1];
  const float* q_w  = (const float*)d_in[2];
  const float* q_b  = (const float*)d_in[3];
  const float* k_w  = (const float*)d_in[4];
  const float* k_b  = (const float*)d_in[5];
  const float* v_w  = (const float*)d_in[6];
  const float* v_b  = (const float*)d_in[7];
  const float* o_w  = (const float*)d_in[8];
  const float* o_b  = (const float*)d_in[9];

  const int M = NB * SEQ;                  // 4096
  char* ws = (char*)d_ws;
  size_t off = 0;
  bfr* xb = (bfr*)(ws + off); off += (size_t)M * HID * 2;      // 16 MB
  bfr* wq = (bfr*)(ws + off); off += (size_t)HID * HID * 2;    //  8 MB
  bfr* wk = (bfr*)(ws + off); off += (size_t)KVD * HID * 2;    //  2 MB
  bfr* wv = (bfr*)(ws + off); off += (size_t)KVD * HID * 2;    //  2 MB
  bfr* wo = (bfr*)(ws + off); off += (size_t)HID * HID * 2;    //  8 MB
  bfr* qb = (bfr*)(ws + off); off += (size_t)M * HID * 2;      // 16 MB
  bfr* kb = (bfr*)(ws + off); off += (size_t)M * KVD * 2;      //  4 MB
  bfr* vT = (bfr*)(ws + off); off += (size_t)M * KVD * 2;      //  4 MB
  bfr* ab = (bfr*)(ws + off); off += (size_t)M * HID * 2;      // 16 MB

  auto cvt = [&](const float* s, bfr* d, int n) {
    cvt_f32_bf16<<<(n + 255) / 256, 256, 0, stream>>>(s, d, n);
  };
  cvt(x,   xb, M * HID);
  cvt(q_w, wq, HID * HID);
  cvt(k_w, wk, KVD * HID);
  cvt(v_w, wv, KVD * HID);
  cvt(o_w, wo, HID * HID);

  // Q/K/V projections
  gemm_bf16<<<dim3(M / BM, HID / BN), 256, 0, stream>>>(
      xb, wq, q_b, qb, M, HID, HID, /*mode=*/0, SEQ);
  gemm_bf16<<<dim3(M / BM, KVD / BN), 256, 0, stream>>>(
      xb, wk, k_b, kb, M, KVD, HID, /*mode=*/0, SEQ);
  gemm_bf16<<<dim3(M / BM, KVD / BN), 256, 0, stream>>>(
      xb, wv, v_b, vT, M, KVD, HID, /*mode=*/2, SEQ);

  // attention: 1 block = 4 q-heads of one KV head on 16 query rows
  attn_kernel<<<dim3(SEQ / 16, NB * NKVH), 128, 0, stream>>>(qb, kb, vT, am, ab);

  // output projection -> f32
  gemm_bf16<<<dim3(M / BM, HID / BN), 256, 0, stream>>>(
      ab, wo, o_b, d_out, M, HID, HID, /*mode=*/1, SEQ);
}
